// HGT_36344013259082
// MI455X (gfx1250) — matrix operand order repaired and checked by reading, not compiled, not run on hardware
//
#include <hip/hip_runtime.h>
#include <stdint.h>

typedef __attribute__((ext_vector_type(16))) _Float16 v16h;
typedef __attribute__((ext_vector_type(4)))  _Float16 v4h;
typedef __attribute__((ext_vector_type(8)))  float    v8f;

#define Hh 8
#define Dd 16
#define Ff 128

// ---------------- helpers ----------------
__device__ __forceinline__ unsigned fkey(float f) {
    unsigned b = __float_as_uint(f);
    return (b & 0x80000000u) ? ~b : (b | 0x80000000u);
}
__device__ __forceinline__ float funkey(unsigned k) {
    unsigned b = (k & 0x80000000u) ? (k & 0x7FFFFFFFu) : ~k;
    return __uint_as_float(b);
}
__device__ __forceinline__ float gelu_exact(float x) {
    return 0.5f * x * (1.0f + erff(x * 0.70710678118654752f));
}

// Packed B fragment layout: half index = ((ks*8 + ntile)*32 + lane)*16 + j
// holding B[k = ks*32 + j + 16*(lane>>4), n = ntile*16 + (lane&15)]

// ---------------- weight prep ----------------
// straight f32 -> f16 convert of a 128x128 weight into packed fragment order
__global__ void k_convert_w(const float* __restrict__ W, _Float16* __restrict__ O) {
    int i = blockIdx.x * blockDim.x + threadIdx.x;
    if (i >= Ff * Ff) return;
    int j    = i & 15;
    int lane = (i >> 4) & 31;
    int tile = (i >> 9) & 7;
    int ks   = (i >> 12) & 3;
    int g = lane >> 4, ml = lane & 15;
    int n = tile * 16 + ml;
    int k = ks * 32 + j + 16 * g;
    O[i] = (_Float16)W[k * Ff + n];
}

// fused weight: F[k, h*16+eo] = sum_d W[k, h*16+d] * rel[h, d, eo]  -> packed f16
// block 128 computes fused bias into Bo (f32).
__global__ __launch_bounds__(128) void k_fuse_w(const float* __restrict__ W,
                                                const float* __restrict__ b,
                                                const float* __restrict__ rel,
                                                _Float16* __restrict__ O,
                                                float* __restrict__ Bo) {
    int c = threadIdx.x;            // output column 0..127
    int h = c >> 4, eo = c & 15;
    int f = blockIdx.x;             // 0..127 rows (k), 128 == bias
    const float* rp = rel + h * 256 + eo;   // rel[h, d, eo], stride 16 over d
    if (f < Ff) {
        const float* wp = W + f * Ff + (h << 4);
        float acc = 0.f;
#pragma unroll
        for (int d = 0; d < 16; ++d) acc += wp[d] * rp[d * 16];
        // packed index for (k=f, n=c)
        int ks = f >> 5, kr = f & 31;
        int g = kr >> 4, j = kr & 15;
        int tile = c >> 4, ml = c & 15;
        int lane = g * 16 + ml;
        O[(((size_t)ks * 8 + tile) * 32 + lane) * 16 + j] = (_Float16)acc;
    } else {
        const float* bp = b + (h << 4);
        float acc = 0.f;
#pragma unroll
        for (int d = 0; d < 16; ++d) acc += bp[d] * rp[d * 16];
        Bo[c] = acc;
    }
}

// ---------------- WMMA GEMM: C[M,128] = epilogue(A[M,128] @ B[128,128]) ----------------
// A: f32 (optionally GELU'd while staging), Bp: f16 in packed fragment order
// epilogue: +bias, optional skip-gate blend with skipX, optional ReLU
__global__ __launch_bounds__(256) void k_gemm128(const float* __restrict__ A,
                                                 const v16h* __restrict__ Bp,
                                                 const float* __restrict__ bias,
                                                 float* __restrict__ C, int M,
                                                 int geluA,
                                                 const float* __restrict__ skipGate,
                                                 const float* __restrict__ skipX,
                                                 int relu) {
    __shared__ _Float16 sA[64 * 32];    // 4 KB
    const int tid  = threadIdx.x;
    const int lane = tid & 31;
    const int wave = tid >> 5;
    const int row0 = blockIdx.x * 64;
    const int ml   = lane & 15;
    const int g    = lane >> 4;

    v8f acc[4];
#pragma unroll
    for (int r = 0; r < 4; ++r) acc[r] = (v8f){0.f,0.f,0.f,0.f,0.f,0.f,0.f,0.f};

    // hoist all four K-step B fragments (each v16h -> 2x global_load_b128)
    v16h fb[4];
#pragma unroll
    for (int ks = 0; ks < 4; ++ks)
        fb[ks] = Bp[((size_t)ks * 8 + wave) * 32 + lane];

    const bool interior = (row0 + 64 <= M);

#pragma unroll
    for (int ks = 0; ks < 4; ++ks) {
        // stage A 64x32 (f32 -> f16, optional gelu) via float4 loads
#pragma unroll
        for (int u = 0; u < 2; ++u) {
            int fi = tid + u * 256;             // 0..511 float4 slots
            int r = fi >> 3, c0 = (fi & 7) * 4;
            int gr = row0 + r;
            float4 v4 = make_float4(0.f, 0.f, 0.f, 0.f);
            if (interior || gr < M)
                v4 = *(const float4*)(A + (size_t)gr * Ff + ks * 32 + c0);
            if (geluA) {
                v4.x = gelu_exact(v4.x); v4.y = gelu_exact(v4.y);
                v4.z = gelu_exact(v4.z); v4.w = gelu_exact(v4.w);
            }
            v4h h4 = { (_Float16)v4.x, (_Float16)v4.y, (_Float16)v4.z, (_Float16)v4.w };
            *(v4h*)(&sA[r * 32 + c0]) = h4;
        }
        __syncthreads();

#pragma unroll
        for (int r = 0; r < 4; ++r) {
            // A fragment: contiguous k = 16*g + j, matching packed B
            v16h fa;
#pragma unroll
            for (int j = 0; j < 16; ++j) {
                int kk = j + g * 16;
                fa[j] = sA[(r * 16 + ml) * 32 + kk];
            }
            acc[r] = __builtin_amdgcn_wmma_f32_16x16x32_f16(
                false, fa, false, fb[ks], (short)0, acc[r], false, false);
        }
        __syncthreads();
    }

    // epilogue
    float sg = 1.f, omg = 0.f;
    if (skipGate) { float s = *skipGate; sg = 1.f / (1.f + expf(-s)); omg = 1.f - sg; }
    const int col = wave * 16 + ml;
    const float bv = bias ? bias[col] : 0.f;
#pragma unroll
    for (int r = 0; r < 4; ++r) {
#pragma unroll
        for (int v = 0; v < 8; ++v) {
            int row = row0 + r * 16 + v + 8 * g;
            if (row < M) {
                float o = acc[r][v] + bv;
                if (skipGate) o = sg * o + omg * skipX[(size_t)row * Ff + col];
                if (relu) o = fmaxf(o, 0.f);
                C[(size_t)row * Ff + col] = o;
            }
        }
    }
}

// ---------------- edge phase ----------------
// thread per (edge, head)
__global__ void k_alpha(const int* __restrict__ src, const int* __restrict__ dst,
                        const float* __restrict__ q, const float* __restrict__ krel,
                        float* __restrict__ alpha, unsigned* __restrict__ amaxk,
                        const float* __restrict__ prel, int E) {
    int i = blockIdx.x * blockDim.x + threadIdx.x;
    if (i >= E * Hh) return;
    int e = i >> 3, h = i & 7;
    int s = src[e], d = dst[e];
    const float4* qp = (const float4*)(q    + (size_t)d * Ff + (h << 4));
    const float4* kp = (const float4*)(krel + (size_t)s * Ff + (h << 4));
    float a = 0.f;
#pragma unroll
    for (int t = 0; t < 4; ++t) {
        float4 qv = qp[t], kv = kp[t];
        a += qv.x * kv.x + qv.y * kv.y + qv.z * kv.z + qv.w * kv.w;
    }
    a *= prel[h] * 0.25f;                 // / sqrt(16)
    alpha[i] = a;
    atomicMax(&amaxk[(size_t)d * Hh + h], fkey(a));
}

__global__ void k_expsum(const int* __restrict__ dst, float* __restrict__ alpha,
                         const unsigned* __restrict__ amaxk, float* __restrict__ den,
                         int E) {
    int i = blockIdx.x * blockDim.x + threadIdx.x;
    if (i >= E * Hh) return;
    int e = i >> 3, h = i & 7;
    int d = dst[e];
    float m = funkey(amaxk[(size_t)d * Hh + h]);
    float ex = expf(alpha[i] - m);
    alpha[i] = ex;
    atomicAdd(&den[(size_t)d * Hh + h], ex);
}

__global__ void k_agg(const int* __restrict__ src, const int* __restrict__ dst,
                      const float* __restrict__ alpha, const float* __restrict__ den,
                      const float* __restrict__ vrel, float* __restrict__ agg, int E) {
    int i = blockIdx.x * blockDim.x + threadIdx.x;
    if (i >= E * Hh) return;
    int e = i >> 3, h = i & 7;
    int s = src[e], d = dst[e];
    float w = alpha[i] / (den[(size_t)d * Hh + h] + 1e-16f);
    const float4* vp = (const float4*)(vrel + (size_t)s * Ff + (h << 4));
    float* ap = agg + (size_t)d * Ff + (h << 4);
#pragma unroll
    for (int t = 0; t < 4; ++t) {
        float4 vv = vp[t];
        atomicAdd(&ap[t * 4 + 0], vv.x * w);
        atomicAdd(&ap[t * 4 + 1], vv.y * w);
        atomicAdd(&ap[t * 4 + 2], vv.z * w);
        atomicAdd(&ap[t * 4 + 3], vv.w * w);
    }
}

// ---------------- host ----------------
extern "C" void kernel_launch(void* const* d_in, const int* in_sizes, int n_in,
                              void* d_out, int out_size, void* d_ws, size_t ws_size,
                              hipStream_t stream) {
    (void)n_in; (void)out_size; (void)ws_size;
    const float* xa   = (const float*)d_in[0];
    const float* xb   = (const float*)d_in[1];
    const int*   ei[3]= {(const int*)d_in[2], (const int*)d_in[3], (const int*)d_in[4]};
    const float* Wk   = (const float*)d_in[5];
    const float* bk   = (const float*)d_in[6];
    const float* Wq   = (const float*)d_in[7];
    const float* bq   = (const float*)d_in[8];
    const float* Wv   = (const float*)d_in[9];
    const float* bv   = (const float*)d_in[10];
    const float* Wa   = (const float*)d_in[11];
    const float* ba   = (const float*)d_in[12];
    const float* skp  = (const float*)d_in[13];
    const float* arel = (const float*)d_in[14];
    const float* mrel = (const float*)d_in[15];
    const float* prel = (const float*)d_in[16];

    const int NA = in_sizes[0] / Ff;
    const int NB = in_sizes[1] / Ff;
    const int E  = in_sizes[2] / 2;
    const int Nmax = (NA > NB) ? NA : NB;

    // workspace carve (256B aligned)
    char* p = (char*)d_ws;
    auto alloc = [&](size_t bytes) -> void* {
        void* r = (void*)p; p += (bytes + 255) & ~(size_t)255; return r;
    };
    const size_t NFa = (size_t)NA * Ff * sizeof(float);
    const size_t NFb = (size_t)NB * Ff * sizeof(float);
    const size_t NFm = (size_t)Nmax * Ff * sizeof(float);
    float* xa1   = (float*)alloc(NFa);
    float* xb1   = (float*)alloc(NFb);
    float* qa    = (float*)alloc(NFa);
    float* qb    = (float*)alloc(NFb);
    float* krel  = (float*)alloc(NFm);
    float* vrel  = (float*)alloc(NFm);
    float* agga  = (float*)alloc(NFa);
    float* aggb  = (float*)alloc(NFb);
    float* alphab= (float*)alloc((size_t)E * Hh * sizeof(float));
    unsigned* amaxk = (unsigned*)alloc((size_t)Nmax * Hh * sizeof(unsigned));
    float* denom = (float*)alloc((size_t)Nmax * Hh * sizeof(float));
    _Float16* w16 = (_Float16*)alloc((size_t)20 * Ff * Ff * sizeof(_Float16));
    float* fbias  = (float*)alloc((size_t)12 * Ff * sizeof(float));

    auto W16 = [&](int l, int i) { return w16 + ((size_t)l * 10 + i) * (Ff * Ff); };
    auto BK  = [&](int l, int e) { return fbias + (size_t)(l * 3 + e) * Ff; };
    auto BV  = [&](int l, int e) { return fbias + (size_t)(6 + l * 3 + e) * Ff; };

    // ---- weight prep ----
    const int srcType[3] = {0, 1, 0};
    for (int l = 0; l < 2; ++l) {
        for (int t = 0; t < 2; ++t) {
            k_convert_w<<<64, 256, 0, stream>>>(Wq + (size_t)(l * 2 + t) * Ff * Ff, W16(l, t));
            k_convert_w<<<64, 256, 0, stream>>>(Wa + (size_t)(l * 2 + t) * Ff * Ff, W16(l, 8 + t));
        }
        for (int e = 0; e < 3; ++e) {
            int st = srcType[e];
            k_fuse_w<<<129, 128, 0, stream>>>(Wk + (size_t)(l * 2 + st) * Ff * Ff,
                                              bk + (size_t)(l * 2 + st) * Ff,
                                              arel + (size_t)(l * 3 + e) * 2048,
                                              W16(l, 2 + e), BK(l, e));
            k_fuse_w<<<129, 128, 0, stream>>>(Wv + (size_t)(l * 2 + st) * Ff * Ff,
                                              bv + (size_t)(l * 2 + st) * Ff,
                                              mrel + (size_t)(l * 3 + e) * 2048,
                                              W16(l, 5 + e), BV(l, e));
        }
    }

    auto gemm = [&](const float* A, const _Float16* B, const float* bias, float* C,
                    int M, int gelu, const float* sgate, const float* sx, int relu) {
        k_gemm128<<<dim3((M + 63) / 64), 256, 0, stream>>>(A, (const v16h*)B, bias, C,
                                                           M, gelu, sgate, sx, relu);
    };

    // ---- layers ----
    const float* xin_a = xa;
    const float* xin_b = xb;
    const int specS[3] = {0, 1, 0};
    const int specD[3] = {1, 0, 0};
    for (int l = 0; l < 2; ++l) {
        hipMemsetAsync(agga, 0, NFa, stream);
        hipMemsetAsync(aggb, 0, NFb, stream);
        gemm(xin_a, W16(l, 0), bq + (size_t)(l * 2 + 0) * Ff, qa, NA, 0, nullptr, nullptr, 0);
        gemm(xin_b, W16(l, 1), bq + (size_t)(l * 2 + 1) * Ff, qb, NB, 0, nullptr, nullptr, 0);

        for (int e = 0; e < 3; ++e) {
            const float* xs = specS[e] ? xin_b : xin_a;
            int Ns = specS[e] ? NB : NA;
            int Nd = specD[e] ? NB : NA;
            float* qd   = specD[e] ? qb : qa;
            float* aggd = specD[e] ? aggb : agga;
            gemm(xs, W16(l, 2 + e), BK(l, e), krel, Ns, 0, nullptr, nullptr, 0);
            gemm(xs, W16(l, 5 + e), BV(l, e), vrel, Ns, 0, nullptr, nullptr, 0);
            hipMemsetAsync(amaxk, 0, (size_t)Nd * Hh * sizeof(unsigned), stream);
            hipMemsetAsync(denom, 0, (size_t)Nd * Hh * sizeof(float), stream);
            int total = E * Hh;
            int blks = (total + 255) / 256;
            const int* s = ei[e];
            const int* d = ei[e] + E;
            k_alpha<<<blks, 256, 0, stream>>>(s, d, qd, krel, alphab, amaxk,
                                              prel + (size_t)(l * 3 + e) * Hh, E);
            k_expsum<<<blks, 256, 0, stream>>>(d, alphab, amaxk, denom, E);
            k_agg<<<blks, 256, 0, stream>>>(s, d, alphab, denom, vrel, aggd, E);
        }

        float* outA = (l == 0) ? xa1 : (float*)d_out;
        float* outB = (l == 0) ? xb1 : ((float*)d_out + (size_t)NA * Ff);
        gemm(agga, W16(l, 8), ba + (size_t)(l * 2 + 0) * Ff, outA, NA, 1, skp + l * 2 + 0, xin_a, l == 0);
        gemm(aggb, W16(l, 9), ba + (size_t)(l * 2 + 1) * Ff, outB, NB, 1, skp + l * 2 + 1, xin_b, l == 0);
        xin_a = xa1;
        xin_b = xb1;
    }
}
